// DualEncoderGraphTransformer_5188320493831
// MI455X (gfx1250) — matrix operand order repaired and checked by reading, not compiled, not run on hardware
//
#include <hip/hip_runtime.h>
#include <hip/hip_fp16.h>

// ---------------- problem constants (from reference) ----------------
#define N_NODES 50000
#define N_EDGES 800000
#define IN_C    300
#define IN_CP   320      // K padded to multiple of 32 for WMMA
#define HID     64
#define HEADS   2
#define OUT_C   2
#define NG      256
#define W1DIM   (HID*HEADS)   // 128

typedef __attribute__((ext_vector_type(16))) _Float16 v16h;
typedef __attribute__((ext_vector_type(8)))  _Float16 v8h;
typedef __attribute__((ext_vector_type(8)))  float    v8f;

static inline int ceildiv(int a, int b) { return (a + b - 1) / b; }

__device__ inline v8f wmma_f16(v16h a, v16h b, v8f c) {
    return __builtin_amdgcn_wmma_f32_16x16x32_f16(false, a, false, b,
                                                  (short)0, c, false, false);
}

// ---------------- small utility kernels ----------------
__global__ void fill_f32(float* __restrict__ p, int n, float v) {
    int i = blockIdx.x * blockDim.x + threadIdx.x;
    if (i < n) p[i] = v;
}

// f32 [sr,sc] -> f16 [dr,dc], zero padded (used for x only)
__global__ void cvt_pad_f16(const float* __restrict__ s, int sr, int sc,
                            _Float16* __restrict__ d, int dr, int dc) {
    int i = blockIdx.x * blockDim.x + threadIdx.x;
    if (i >= dr * dc) return;
    int r = i / dc, c = i - r * dc;
    float v = (r < sr && c < sc) ? s[r * sc + c] : 0.0f;
    d[i] = (_Float16)v;
}

// Pack B [Krows, Nn] f32 row-major into WMMA-fragment-ordered f16.
// Fragment layout per ISA 7.12.2 (B 32x16, 16-bit):
//   lane l: col n = l&15, group g = l>>4;  element e -> K = kt*32 + (e&7) + 8*g + 16*(e>>3)
// Packed address: dst[(((nt*Ktiles)+kt)*32 + l)*16 + e]  (32B contiguous per lane)
__global__ void pack_b_f16(const float* __restrict__ B, int Krows, int Nn,
                           _Float16* __restrict__ dst, int Ktiles) {
    int i = blockIdx.x * blockDim.x + threadIdx.x;
    int total = Ktiles * 32 * 16 * (Nn / 16);
    if (i >= total) return;
    int e  = i & 15;
    int l  = (i >> 4) & 31;
    int t  = i >> 9;                 // nt*Ktiles + kt
    int kt = t % Ktiles, nt = t / Ktiles;
    int krow = kt * 32 + (e & 7) + 8 * (l >> 4) + 16 * (e >> 3);
    int col  = nt * 16 + (l & 15);
    float v = (krow < Krows) ? B[(size_t)krow * Nn + col] : 0.0f;
    dst[i] = (_Float16)v;
}

__device__ inline float atomicMaxF(float* addr, float val) {
    int* ai = (int*)addr;
    if (__float_as_int(val) >= 0)
        return __int_as_float(atomicMax(ai, __float_as_int(val)));
    return __int_as_float((int)atomicMin((unsigned int*)ai,
                                         (unsigned int)__float_as_int(val)));
}

// ---------------- WMMA GEMM: C[M, Ngrp*64] = relu?(A[M,K]*B + bias) ----------------
// One wave computes a 32x64 strip of C: 2 M-subtiles x 4 N-subtiles = 8 accumulators.
// Per K-step: 4 A b128 loads + 8 B b128 loads feed 8 WMMAs (B reused across M).
// A fragment (16-bit 16x32, ISA 7.12.2): lane m=lane&15, g=lane>>4;
//   elems 0..7 -> K=k0+8g.., elems 8..15 -> K=k0+16+8g..  (two 16B loads)
// B fragments come from pack_b_f16: one contiguous 32B v16h load each.
__global__ void __launch_bounds__(128)
gemm_wmma_f16(const _Float16* __restrict__ A, int lda,
              const _Float16* __restrict__ Bp,
              const float* __restrict__ bias,
              float* __restrict__ Cf, _Float16* __restrict__ Ch, int ldc,
              int Mtiles, int Ngrp, int Ktiles, int relu)
{
    int wave = threadIdx.x >> 5;
    int lane = threadIdx.x & 31;
    int Mpairs = (Mtiles + 1) >> 1;
    int job  = blockIdx.x * 4 + wave;       // wave-uniform -> EXEC all-1s at WMMA
    if (job >= Mpairs * Ngrp) return;
    int mp = job % Mpairs, ng = job / Mpairs;
    int m0 = mp * 32;
    int valid1 = (mp * 2 + 1) < Mtiles;     // second M-subtile in range?
    int g  = lane >> 4;
    int ln = lane & 15;

    v8f acc00 = {}, acc01 = {}, acc02 = {}, acc03 = {};
    v8f acc10 = {}, acc11 = {}, acc12 = {}, acc13 = {};
    const _Float16* Arow0 = A + (size_t)(m0 + ln) * lda + g * 8;
    const _Float16* Arow1 = Arow0 + (size_t)16 * lda;
    const size_t nstride = (size_t)Ktiles * 32 * 16;         // halves per n-tile block
    const _Float16* Bl = Bp + (size_t)(ng * 4) * nstride + (size_t)lane * 16;

    for (int kt = 0; kt < Ktiles; ++kt) {
        __builtin_prefetch(Arow0 + kt * 32 + 64, 0, 1);      // global_prefetch_b8
        union { v16h v; v8h h2[2]; } a0, a1;
        a0.h2[0] = *(const v8h*)(Arow0 + kt * 32);
        a0.h2[1] = *(const v8h*)(Arow0 + kt * 32 + 16);
        a1.h2[0] = *(const v8h*)(Arow1 + kt * 32);
        a1.h2[1] = *(const v8h*)(Arow1 + kt * 32 + 16);
        const _Float16* bk = Bl + (size_t)kt * 32 * 16;
        v16h b0 = *(const v16h*)(bk);
        v16h b1 = *(const v16h*)(bk + nstride);
        v16h b2 = *(const v16h*)(bk + 2 * nstride);
        v16h b3 = *(const v16h*)(bk + 3 * nstride);
        acc00 = wmma_f16(a0.v, b0, acc00);
        acc01 = wmma_f16(a0.v, b1, acc01);
        acc02 = wmma_f16(a0.v, b2, acc02);
        acc03 = wmma_f16(a0.v, b3, acc03);
        acc10 = wmma_f16(a1.v, b0, acc10);
        acc11 = wmma_f16(a1.v, b1, acc11);
        acc12 = wmma_f16(a1.v, b2, acc12);
        acc13 = wmma_f16(a1.v, b3, acc13);
    }

#pragma unroll
    for (int s = 0; s < 4; ++s) {
        int n0 = (ng * 4 + s) * 16;
        float bv = bias ? bias[n0 + ln] : 0.0f;
#pragma unroll
        for (int mh = 0; mh < 2; ++mh) {
            if (mh == 1 && !valid1) continue;               // wave-uniform guard
            v8f acc;
            if (mh == 0) acc = s == 0 ? acc00 : s == 1 ? acc01 : s == 2 ? acc02 : acc03;
            else         acc = s == 0 ? acc10 : s == 1 ? acc11 : s == 2 ? acc12 : acc13;
#pragma unroll
            for (int j = 0; j < 8; ++j) {
                int row = m0 + mh * 16 + j + g * 8;
                float cv = acc[j] + bv;
                if (relu) cv = cv > 0.0f ? cv : 0.0f;
                size_t idx = (size_t)row * ldc + n0 + ln;
                if (Cf) Cf[idx] = cv;
                if (Ch) Ch[idx] = (_Float16)cv;
            }
        }
    }
}

// ---------------- edge attention passes (one wave per edge) ----------------
template<int WIDTH, int NH>
__global__ void __launch_bounds__(256)
edge_scores(const int* __restrict__ src, const int* __restrict__ dst,
            const _Float16* __restrict__ q, const _Float16* __restrict__ k,
            float* __restrict__ alpha, float* __restrict__ amax, int E)
{
    int lane = threadIdx.x & 31;
    int e = (blockIdx.x * blockDim.x + threadIdx.x) >> 5;
    if (e >= E) return;
    int s = src[e], d = dst[e];
    const int PER = WIDTH / 32;        // dims per lane (4 or 2)
    const int GRP = 32 / NH;           // lanes per head
    int d0 = lane * PER;
    const _Float16* qp = q + (size_t)d * WIDTH + d0;
    const _Float16* kp = k + (size_t)s * WIDTH + d0;
    float part = 0.0f;
#pragma unroll
    for (int t = 0; t < PER; ++t)
        part += (float)qp[t] * (float)kp[t];
#pragma unroll
    for (int m = GRP >> 1; m >= 1; m >>= 1) part += __shfl_xor(part, m, 32);
    int h = lane / GRP;
    if ((lane & (GRP - 1)) == 0) {
        float a = part * 0.125f;       // 1/sqrt(64)
        alpha[(size_t)e * NH + h] = a;
        atomicMaxF(&amax[(size_t)d * NH + h], a);
    }
}

__global__ void edge_softmax_sum(const int* __restrict__ dst,
                                 float* __restrict__ alpha,
                                 const float* __restrict__ amax,
                                 float* __restrict__ asum, int E, int NH)
{
    int i = blockIdx.x * blockDim.x + threadIdx.x;
    if (i >= E * NH) return;
    int e = i / NH, h = i - e * NH;
    int d = dst[e];
    float ea = __expf(alpha[i] - amax[(size_t)d * NH + h]);
    alpha[i] = ea;
    atomicAdd(&asum[(size_t)d * NH + h], ea);
}

template<int WIDTH, int NH>
__global__ void __launch_bounds__(256)
edge_scatter(const int* __restrict__ src, const int* __restrict__ dst,
             const _Float16* __restrict__ v, const float* __restrict__ alpha,
             const float* __restrict__ asum, float* __restrict__ agg, int E)
{
    int lane = threadIdx.x & 31;
    int e = (blockIdx.x * blockDim.x + threadIdx.x) >> 5;
    if (e >= E) return;
    int s = src[e], d = dst[e];
    const int PER = WIDTH / 32;
    const int GRP = 32 / NH;
    int h = lane / GRP;
    float w = alpha[(size_t)e * NH + h] /
              (asum[(size_t)d * NH + h] + 1e-16f);
    int d0 = lane * PER;
    const _Float16* vp = v + (size_t)s * WIDTH + d0;
    float* ap = agg + (size_t)d * WIDTH + d0;
#pragma unroll
    for (int t = 0; t < PER; ++t)
        atomicAdd(&ap[t], w * (float)vp[t]);
}

// ---------------- combine / pool / head ----------------
__global__ void combine_relu_f16(const float* __restrict__ a, const float* __restrict__ b,
                                 _Float16* __restrict__ out, int n) {
    int i = blockIdx.x * blockDim.x + threadIdx.x;
    if (i >= n) return;
    float v = a[i] + b[i];
    out[i] = (_Float16)(v > 0.0f ? v : 0.0f);
}

__global__ void combine_f32(const float* __restrict__ a, const float* __restrict__ b,
                            float* __restrict__ out, int n) {
    int i = blockIdx.x * blockDim.x + threadIdx.x;
    if (i >= n) return;
    out[i] = a[i] + b[i];
}

__global__ void pool_accum(const float* __restrict__ h, const int* __restrict__ batch,
                           float* __restrict__ gsum, int n) {   // n = N*64
    int i = blockIdx.x * blockDim.x + threadIdx.x;
    if (i >= n) return;
    int node = i >> 6, c = i & 63;
    atomicAdd(&gsum[batch[node] * 64 + c], h[i]);
}

__global__ void pool_count(const int* __restrict__ batch, float* __restrict__ gcnt, int n) {
    int i = blockIdx.x * blockDim.x + threadIdx.x;
    if (i >= n) return;
    atomicAdd(&gcnt[batch[i]], 1.0f);
}

__global__ void __launch_bounds__(64)
mlp_head(const float* __restrict__ gsum, const float* __restrict__ gcnt,
         const float* __restrict__ W1p, const float* __restrict__ b1,
         const float* __restrict__ W2p, const float* __restrict__ b2,
         float* __restrict__ out)
{
    __shared__ float pooled[64];
    __shared__ float y1[64];
    int g = blockIdx.x, c = threadIdx.x;
    float cnt = gcnt[g];
    cnt = cnt > 1.0f ? cnt : 1.0f;
    pooled[c] = gsum[g * 64 + c] / cnt;
    __syncthreads();
    float acc = b1[c];
    for (int kk = 0; kk < 64; ++kk) acc += pooled[kk] * W1p[kk * 64 + c];
    y1[c] = acc > 0.0f ? acc : 0.0f;
    __syncthreads();
    if (c < OUT_C) {
        float o = b2[c];
        for (int kk = 0; kk < 64; ++kk) o += y1[kk] * W2p[kk * OUT_C + c];
        out[g * OUT_C + c] = o;
    }
}

// ---------------- host orchestration ----------------
extern "C" void kernel_launch(void* const* d_in, const int* in_sizes, int n_in,
                              void* d_out, int out_size, void* d_ws, size_t ws_size,
                              hipStream_t stream)
{
    (void)in_sizes; (void)n_in; (void)out_size; (void)ws_size;

    const float* x     = (const float*)d_in[0];
    const int*   ei    = (const int*)d_in[1];
    const int*   batch = (const int*)d_in[2];
    const float* syn_W = (const float*)d_in[3];  const float* syn_b = (const float*)d_in[4];
    const float* ant_W = (const float*)d_in[5];  const float* ant_b = (const float*)d_in[6];
    const float* fus_W = (const float*)d_in[7];  const float* fus_b = (const float*)d_in[8];
    const float* c1_Wq = (const float*)d_in[9];  const float* c1_bq = (const float*)d_in[10];
    const float* c1_Wk = (const float*)d_in[11]; const float* c1_bk = (const float*)d_in[12];
    const float* c1_Wv = (const float*)d_in[13]; const float* c1_bv = (const float*)d_in[14];
    const float* c1_Ws = (const float*)d_in[15]; const float* c1_bs = (const float*)d_in[16];
    const float* c2_Wq = (const float*)d_in[17]; const float* c2_bq = (const float*)d_in[18];
    const float* c2_Wk = (const float*)d_in[19]; const float* c2_bk = (const float*)d_in[20];
    const float* c2_Wv = (const float*)d_in[21]; const float* c2_bv = (const float*)d_in[22];
    const float* c2_Ws = (const float*)d_in[23]; const float* c2_bs = (const float*)d_in[24];
    const float* l1_W  = (const float*)d_in[25]; const float* l1_b  = (const float*)d_in[26];
    const float* l2_W  = (const float*)d_in[27]; const float* l2_b  = (const float*)d_in[28];

    const int* src = ei;
    const int* dst = ei + N_EDGES;

    // ---- workspace carve-out (256B aligned segments) ----
    char* ws = (char*)d_ws;
    size_t off = 0;
    auto take = [&](size_t bytes) -> char* {
        off = (off + 255) & ~(size_t)255;
        char* p = ws + off; off += bytes; return p;
    };
    _Float16* xh   = (_Float16*)take((size_t)N_NODES * IN_CP * 2);  // padded x, f16
    _Float16* fi   = (_Float16*)take((size_t)N_NODES * W1DIM * 2);  // [syn|ant] relu'd
    _Float16* xf   = (_Float16*)take((size_t)N_NODES * W1DIM * 2);  // x_fused
    _Float16* q1   = (_Float16*)take((size_t)N_NODES * W1DIM * 2);
    _Float16* k1   = (_Float16*)take((size_t)N_NODES * W1DIM * 2);
    _Float16* v1   = (_Float16*)take((size_t)N_NODES * W1DIM * 2);
    float*    sk1  = (float*)take((size_t)N_NODES * W1DIM * 4);
    float*    agg1 = (float*)take((size_t)N_NODES * W1DIM * 4);
    _Float16* h1   = (_Float16*)take((size_t)N_NODES * W1DIM * 2);
    _Float16* q2   = (_Float16*)take((size_t)N_NODES * HID * 2);
    _Float16* k2   = (_Float16*)take((size_t)N_NODES * HID * 2);
    _Float16* v2   = (_Float16*)take((size_t)N_NODES * HID * 2);
    float*    sk2  = (float*)take((size_t)N_NODES * HID * 4);
    float*    agg2 = (float*)take((size_t)N_NODES * HID * 4);
    float*    h2   = (float*)take((size_t)N_NODES * HID * 4);
    float*    amax = (float*)take((size_t)N_NODES * HEADS * 4);
    float*    asum = (float*)take((size_t)N_NODES * HEADS * 4);
    float*    alpha= (float*)take((size_t)N_EDGES * HEADS * 4);
    float*    gsum = (float*)take((size_t)NG * HID * 4);
    float*    gcnt = (float*)take((size_t)NG * 4);
    // packed f16 weights (fragment-ordered)
    _Float16* synWp = (_Float16*)take((size_t)IN_CP * HID * 2);
    _Float16* antWp = (_Float16*)take((size_t)IN_CP * HID * 2);
    _Float16* fusWp = (_Float16*)take((size_t)W1DIM * W1DIM * 2);
    _Float16* WqA   = (_Float16*)take((size_t)W1DIM * W1DIM * 2);
    _Float16* WkA   = (_Float16*)take((size_t)W1DIM * W1DIM * 2);
    _Float16* WvA   = (_Float16*)take((size_t)W1DIM * W1DIM * 2);
    _Float16* WsA   = (_Float16*)take((size_t)W1DIM * W1DIM * 2);
    _Float16* WqB   = (_Float16*)take((size_t)W1DIM * HID * 2);
    _Float16* WkB   = (_Float16*)take((size_t)W1DIM * HID * 2);
    _Float16* WvB   = (_Float16*)take((size_t)W1DIM * HID * 2);
    _Float16* WsB   = (_Float16*)take((size_t)W1DIM * HID * 2);
    // slack so phantom-tile A reads of the last M-pair stay inside the workspace
    (void)take((size_t)32 * IN_CP * 2);

    const int T = 256;
    const int KT_IN = IN_CP / 32;   // 10
    const int KT_W1 = W1DIM / 32;   // 4

    auto pack = [&](const float* B, int Krows, int Nn, _Float16* dstp, int Ktiles) {
        int n = Ktiles * 32 * 16 * (Nn / 16);
        pack_b_f16<<<ceildiv(n, T), T, 0, stream>>>(B, Krows, Nn, dstp, Ktiles);
    };

    // ---- stage 0: conversions / weight packing ----
    cvt_pad_f16<<<ceildiv(N_NODES * IN_CP, T), T, 0, stream>>>(
        x, N_NODES, IN_C, xh, N_NODES, IN_CP);
    pack(syn_W, IN_C, HID, synWp, KT_IN);
    pack(ant_W, IN_C, HID, antWp, KT_IN);
    pack(fus_W, W1DIM, W1DIM, fusWp, KT_W1);
    pack(c1_Wq, W1DIM, W1DIM, WqA, KT_W1);
    pack(c1_Wk, W1DIM, W1DIM, WkA, KT_W1);
    pack(c1_Wv, W1DIM, W1DIM, WvA, KT_W1);
    pack(c1_Ws, W1DIM, W1DIM, WsA, KT_W1);
    pack(c2_Wq, W1DIM, HID, WqB, KT_W1);
    pack(c2_Wk, W1DIM, HID, WkB, KT_W1);
    pack(c2_Wv, W1DIM, HID, WvB, KT_W1);
    pack(c2_Ws, W1DIM, HID, WsB, KT_W1);

    const int Mt = N_NODES / 16;          // 3125, exact
    const int Mpairs = (Mt + 1) / 2;      // 1563 (last pair half-valid)
    auto gemm = [&](const _Float16* A, int lda, const _Float16* Bp,
                    const float* bias, float* Cf, _Float16* Ch, int ldc,
                    int Ngrp, int Ktiles, int relu) {
        int jobs = Mpairs * Ngrp;
        gemm_wmma_f16<<<ceildiv(jobs, 4), 128, 0, stream>>>(
            A, lda, Bp, bias, Cf, Ch, ldc, Mt, Ngrp, Ktiles, relu);
    };

    // ---- stage 1: dual encoders + fusion ----
    gemm(xh, IN_CP, synWp, syn_b, nullptr, fi,       W1DIM, 1, KT_IN, 1);
    gemm(xh, IN_CP, antWp, ant_b, nullptr, fi + HID, W1DIM, 1, KT_IN, 1);
    gemm(fi, W1DIM, fusWp, fus_b, nullptr, xf, W1DIM, 2, KT_W1, 0);

    // ---- stage 2: conv1 projections ----
    gemm(xf, W1DIM, WqA, c1_bq, nullptr, q1, W1DIM, 2, KT_W1, 0);
    gemm(xf, W1DIM, WkA, c1_bk, nullptr, k1, W1DIM, 2, KT_W1, 0);
    gemm(xf, W1DIM, WvA, c1_bv, nullptr, v1, W1DIM, 2, KT_W1, 0);
    gemm(xf, W1DIM, WsA, c1_bs, sk1, nullptr, W1DIM, 2, KT_W1, 0);

    // ---- conv1 edge softmax-attention ----
    fill_f32<<<ceildiv(N_NODES * HEADS, T), T, 0, stream>>>(amax, N_NODES * HEADS, -3.4e38f);
    fill_f32<<<ceildiv(N_NODES * HEADS, T), T, 0, stream>>>(asum, N_NODES * HEADS, 0.0f);
    fill_f32<<<ceildiv(N_NODES * W1DIM, T), T, 0, stream>>>(agg1, N_NODES * W1DIM, 0.0f);
    edge_scores<W1DIM, HEADS><<<ceildiv(N_EDGES * 32, T), T, 0, stream>>>(
        src, dst, q1, k1, alpha, amax, N_EDGES);
    edge_softmax_sum<<<ceildiv(N_EDGES * HEADS, T), T, 0, stream>>>(
        dst, alpha, amax, asum, N_EDGES, HEADS);
    edge_scatter<W1DIM, HEADS><<<ceildiv(N_EDGES * 32, T), T, 0, stream>>>(
        src, dst, v1, alpha, asum, agg1, N_EDGES);
    combine_relu_f16<<<ceildiv(N_NODES * W1DIM, T), T, 0, stream>>>(
        agg1, sk1, h1, N_NODES * W1DIM);

    // ---- stage 3: conv2 (heads=1, d=64) ----
    gemm(h1, W1DIM, WqB, c2_bq, nullptr, q2, HID, 1, KT_W1, 0);
    gemm(h1, W1DIM, WkB, c2_bk, nullptr, k2, HID, 1, KT_W1, 0);
    gemm(h1, W1DIM, WvB, c2_bv, nullptr, v2, HID, 1, KT_W1, 0);
    gemm(h1, W1DIM, WsB, c2_bs, sk2, nullptr, HID, 1, KT_W1, 0);

    fill_f32<<<ceildiv(N_NODES, T), T, 0, stream>>>(amax, N_NODES, -3.4e38f);
    fill_f32<<<ceildiv(N_NODES, T), T, 0, stream>>>(asum, N_NODES, 0.0f);
    fill_f32<<<ceildiv(N_NODES * HID, T), T, 0, stream>>>(agg2, N_NODES * HID, 0.0f);
    edge_scores<HID, 1><<<ceildiv(N_EDGES * 32, T), T, 0, stream>>>(
        src, dst, q2, k2, alpha, amax, N_EDGES);
    edge_softmax_sum<<<ceildiv(N_EDGES, T), T, 0, stream>>>(
        dst, alpha, amax, asum, N_EDGES, 1);
    edge_scatter<HID, 1><<<ceildiv(N_EDGES * 32, T), T, 0, stream>>>(
        src, dst, v2, alpha, asum, agg2, N_EDGES);
    combine_f32<<<ceildiv(N_NODES * HID, T), T, 0, stream>>>(
        agg2, sk2, h2, N_NODES * HID);

    // ---- stage 4: global mean pool + MLP head ----
    fill_f32<<<ceildiv(NG * HID, T), T, 0, stream>>>(gsum, NG * HID, 0.0f);
    fill_f32<<<1, NG, 0, stream>>>(gcnt, NG, 0.0f);
    pool_accum<<<ceildiv(N_NODES * HID, T), T, 0, stream>>>(h2, batch, gsum, N_NODES * HID);
    pool_count<<<ceildiv(N_NODES, T), T, 0, stream>>>(batch, gcnt, N_NODES);
    mlp_head<<<NG, 64, 0, stream>>>(gsum, gcnt, l1_W, l1_b, l2_W, l2_b, (float*)d_out);
}
